// policy_nn_attn_292057776602
// MI455X (gfx1250) — matrix operand order repaired and checked by reading, not compiled, not run on hardware
//
#include <hip/hip_runtime.h>
#include <hip/hip_bf16.h>

// ---------------------------------------------------------------------------
// Problem constants (from reference)
// ---------------------------------------------------------------------------
#define BROWS 8192
#define DEG   32
#define DDIM  256
#define ADIM  128
#define H0    768     // 2*D + D (concat width)
#define H1DIM 512
#define H2DIM 1024
#define H3DIM 512
#define SLOPE 0.2f

typedef __attribute__((ext_vector_type(16))) __bf16 v16bf;
typedef __attribute__((ext_vector_type(8)))  __bf16 v8bf;
typedef __attribute__((ext_vector_type(8)))  float  v8f;
typedef __attribute__((ext_vector_type(4)))  int    v4i;

// CDNA5 async global->LDS copy path (guarded; falls back to plain copies)
#if defined(__AMDGCN__) && __has_builtin(__builtin_amdgcn_global_load_async_to_lds_b128) && __has_builtin(__builtin_amdgcn_s_wait_asynccnt)
#define HAVE_ASYNC_LDS 1
#else
#define HAVE_ASYNC_LDS 0
#endif

#define AS1 __attribute__((address_space(1)))
#define AS3 __attribute__((address_space(3)))

// float -> bf16 round-to-nearest-even (bit trick; avoids relying on __bf16 cvt lowering)
__device__ __forceinline__ __bf16 f2bf(float f) {
    unsigned u = __builtin_bit_cast(unsigned, f);
    unsigned r = u + 0x7FFFu + ((u >> 16) & 1u);
    unsigned short h = (unsigned short)(r >> 16);
    return __builtin_bit_cast(__bf16, h);
}

// ---------------------------------------------------------------------------
// Kernel 1: fold attn_w @ w_now / w_nb into 256-vectors + scalar constant.
// scores = now_emb.vnow + neigh.vnb + (attn_b.(w_now+w_nb) + aw_b)
// ---------------------------------------------------------------------------
__global__ void prep_attn_kernel(const float* __restrict__ attn_w,
                                 const float* __restrict__ attn_b,
                                 const float* __restrict__ aw_w,
                                 const float* __restrict__ aw_b,
                                 float* __restrict__ vnow,
                                 float* __restrict__ vnb,
                                 float* __restrict__ ctot) {
    int d = threadIdx.x;   // 256 threads, one block
    float sn = 0.f, sb = 0.f;
    const float* wrow = attn_w + (size_t)d * ADIM;
    for (int a = 0; a < ADIM; ++a) {
        float w = wrow[a];
        sn += w * aw_w[a];
        sb += w * aw_w[ADIM + a];
    }
    vnow[d] = sn;
    vnb[d]  = sb;
    if (d == 0) {
        float c = aw_b[0];
        for (int a = 0; a < ADIM; ++a) c += attn_b[a] * (aw_w[a] + aw_w[ADIM + a]);
        *ctot = c;
    }
}

// ---------------------------------------------------------------------------
// Kernel 2: W (K x N, fp32 row-major) -> Wt (N x K, bf16 row-major)
// so WMMA B-fragments become contiguous per-lane loads.
// ---------------------------------------------------------------------------
__global__ void conv_wt_kernel(const float* __restrict__ W,
                               __bf16* __restrict__ Wt, int K, int N) {
    int idx = blockIdx.x * blockDim.x + threadIdx.x;
    if (idx >= K * N) return;
    int n = idx / K;
    int k = idx - n * K;
    Wt[idx] = f2bf(W[(size_t)k * N + n]);
}

// ---------------------------------------------------------------------------
// Kernel 3: attention + aggregation + concat. One wave (32 lanes) per row.
// Lane j owns neighbor j for the softmax (wave32 == DEG), and channel chunk
// [8*lane, 8*lane+8) for the dots / aggregation (coalesced float4 loads).
// ---------------------------------------------------------------------------
__global__ void attn_kernel(const float* __restrict__ state,
                            const float* __restrict__ now_emb,
                            const float* __restrict__ neigh,
                            const float* __restrict__ vnow,
                            const float* __restrict__ vnb,
                            const float* __restrict__ ctot,
                            __bf16* __restrict__ X) {
    const int lane = threadIdx.x & 31;
    const int wave = threadIdx.x >> 5;
    const int b = blockIdx.x * 8 + wave;
    if (b >= BROWS) return;

    // lane's 8-channel slice of vnow / vnb
    float wn[8], wb[8];
    {
        const float4* p0 = (const float4*)(vnow + lane * 8);
        const float4* p1 = (const float4*)(vnb  + lane * 8);
        float4 a = p0[0], c = p0[1], e = p1[0], g = p1[1];
        wn[0]=a.x; wn[1]=a.y; wn[2]=a.z; wn[3]=a.w; wn[4]=c.x; wn[5]=c.y; wn[6]=c.z; wn[7]=c.w;
        wb[0]=e.x; wb[1]=e.y; wb[2]=e.z; wb[3]=e.w; wb[4]=g.x; wb[5]=g.y; wb[6]=g.z; wb[7]=g.w;
    }

    // s_now = dot(now_emb[b], vnow)
    float s_now = 0.f;
    {
        const float4* p = (const float4*)(now_emb + (size_t)b * DDIM + lane * 8);
        float4 a = p[0], c = p[1];
        s_now = a.x*wn[0]+a.y*wn[1]+a.z*wn[2]+a.w*wn[3]+c.x*wn[4]+c.y*wn[5]+c.z*wn[6]+c.w*wn[7];
        for (int m = 16; m > 0; m >>= 1) s_now += __shfl_xor(s_now, m);
    }
    const float cadd = *ctot + s_now;

    // scores: whole wave computes dot for neighbor j; lane j keeps it
    float myscore = 0.f;
    const size_t ebase = (size_t)b * DEG;
    for (int j = 0; j < DEG; ++j) {
        const float4* p = (const float4*)(neigh + (ebase + j) * DDIM + lane * 8);
        float4 a = p[0], c = p[1];
        float t = a.x*wb[0]+a.y*wb[1]+a.z*wb[2]+a.w*wb[3]+c.x*wb[4]+c.y*wb[5]+c.z*wb[6]+c.w*wb[7];
        for (int m = 16; m > 0; m >>= 1) t += __shfl_xor(t, m);
        if (lane == j) myscore = t;
    }
    myscore += cadd;
    myscore = (myscore >= 0.f) ? myscore : SLOPE * myscore;   // leaky relu

    // softmax across the 32 lanes
    float mx = myscore;
    for (int m = 16; m > 0; m >>= 1) mx = fmaxf(mx, __shfl_xor(mx, m));
    float e = __expf(myscore - mx);
    float ssum = e;
    for (int m = 16; m > 0; m >>= 1) ssum += __shfl_xor(ssum, m);
    const float alpha = e / ssum;

    // aggregation: lane owns channels [8*lane, 8*lane+8)
    float acc[8] = {0.f,0.f,0.f,0.f,0.f,0.f,0.f,0.f};
    for (int j = 0; j < DEG; ++j) {
        float aj = __shfl(alpha, j);
        const float4* p = (const float4*)(neigh + (ebase + j) * DDIM + lane * 8);
        float4 a = p[0], c = p[1];
        acc[0] += aj*a.x; acc[1] += aj*a.y; acc[2] += aj*a.z; acc[3] += aj*a.w;
        acc[4] += aj*c.x; acc[5] += aj*c.y; acc[6] += aj*c.z; acc[7] += aj*c.w;
    }

    // write X row b (bf16): [ state(512) | agg(256) ]
    __bf16* xr = X + (size_t)b * H0;
    const float* st = state + (size_t)b * (2 * DDIM);
    for (int i = lane; i < 2 * DDIM; i += 32) xr[i] = f2bf(st[i]);
    for (int i = 0; i < 8; ++i) xr[2 * DDIM + lane * 8 + i] = f2bf(acc[i]);
}

// ---------------------------------------------------------------------------
// WMMA fragment loads (wave32, v_wmma_f32_16x16x32_bf16 layouts per ISA 7.12.2)
// A (16x32, row-major src): lane L: row = L&15; khalf=(L>>4)*8;
//   elems 0..7  = K[khalf .. khalf+7], elems 8..15 = K[16+khalf .. 16+khalf+7]
// B (32x16): lane L: col = L&15; same K order as A (khalf, 16+khalf) of that col
// ---------------------------------------------------------------------------
__device__ __forceinline__ v16bf load_afrag(const __bf16* __restrict__ A, int ldk,
                                            int m0, int k0, int lane) {
    int r  = m0 + (lane & 15);
    int kh = (lane >> 4) * 8;
    const __bf16* p = A + (size_t)r * ldk + k0 + kh;
    v8bf lo = *(const v8bf*)p;
    v8bf hi = *(const v8bf*)(p + 16);
    return __builtin_shufflevector(lo, hi, 0,1,2,3,4,5,6,7,8,9,10,11,12,13,14,15);
}

#define BSTR 40   // padded LDS row stride (elements) to spread banks

__device__ __forceinline__ v16bf load_bfrag_lds(const __bf16* base, int n, int kh) {
    const __bf16* p = base + (size_t)n * BSTR + kh;
    v8bf lo = *(const v8bf*)p;
    v8bf hi = *(const v8bf*)(p + 16);
    return __builtin_shufflevector(lo, hi, 0,1,2,3,4,5,6,7,8,9,10,11,12,13,14,15);
}

// ---------------------------------------------------------------------------
// Kernels 4-6: bf16 WMMA GEMM  C[M,N] = act(A[M,K] @ W[K,N] + bias)
// W supplied pre-transposed as Bt[N,K] bf16.
// Block = 8 waves covering 32(M) x 256(N); wave tile 16 x 64 (4 v_wmma/k-step).
// The 32k x 256n B-tile is staged per block into LDS (async global->LDS on
// CDNA5, double buffered, one s_wait_asynccnt + barrier per k-step); A
// fragments are double-buffered in registers. M,N,K multiples of 32/256/32.
// ---------------------------------------------------------------------------
template<bool RELU, bool OUT_BF16>
__global__ __launch_bounds__(256) void gemm_wmma_kernel(const __bf16* __restrict__ A,
                                 const __bf16* __restrict__ Bt,
                                 const float* __restrict__ bias,
                                 void* __restrict__ Cout,
                                 int M, int N, int K) {
    __shared__ __bf16 Bs[2][256 * BSTR];   // 2 x 20 KB

    const int tid  = threadIdx.x;
    const int lane = tid & 31;
    const int wid  = tid >> 5;                 // 0..7
    const int m0   = blockIdx.x * 32 + (wid & 1) * 16;
    const int nblk = blockIdx.y * 256;
    const int n0w  = (wid >> 1) * 64;          // wave's N offset within block tile

    // cooperative B staging: thread t owns column n = nblk + t (32 k-elems = 64B)
    const __bf16* bsrc = Bt + (size_t)(nblk + tid) * K;
    __bf16* bdst[2] = { &Bs[0][tid * BSTR], &Bs[1][tid * BSTR] };

#if HAVE_ASYNC_LDS
#define STAGE_B(buf, k0) do {                                              \
        _Pragma("unroll")                                                  \
        for (int q = 0; q < 4; ++q)                                        \
            __builtin_amdgcn_global_load_async_to_lds_b128(                \
                (AS1 v4i*)(bsrc + (k0) + q * 8),                           \
                (AS3 v4i*)(bdst[buf] + q * 8), 0, 0);                      \
    } while (0)
#define WAIT_ASYNC() __builtin_amdgcn_s_wait_asynccnt(0)
#else
#define STAGE_B(buf, k0) do {                                              \
        _Pragma("unroll")                                                  \
        for (int q = 0; q < 4; ++q)                                        \
            *(v8bf*)(bdst[buf] + q * 8) = *(const v8bf*)(bsrc + (k0) + q * 8); \
    } while (0)
#define WAIT_ASYNC() do {} while (0)
#endif

    // prologue: stage tile 0, load A fragment 0
    STAGE_B(0, 0);
    v16bf af = load_afrag(A, K, m0, 0, lane);

    v8f acc0 = {}, acc1 = {}, acc2 = {}, acc3 = {};
    const int nk = K / 32;
    const int kh = (lane >> 4) * 8;

    for (int kt = 0; kt < nk; ++kt) {
        WAIT_ASYNC();          // this wave's staged tile landed in LDS
        __syncthreads();       // everyone's tile visible; prev buffer free

        const __bf16* bufB = (kt & 1) ? &Bs[1][0] : &Bs[0][0];

        // stage next tile / load next A fragment while computing this one
        if (kt + 1 < nk) {
            STAGE_B((kt & 1) ? 0 : 1, (kt + 1) * 32);
            // prefetch upcoming A rows into cache
            __builtin_prefetch(A + (size_t)(m0 + (lane & 15)) * K + (kt + 1) * 32, 0, 1);
        }
        v16bf afn = af;
        if (kt + 1 < nk) afn = load_afrag(A, K, m0, (kt + 1) * 32, lane);

        v16bf b0 = load_bfrag_lds(bufB, n0w +  0 + (lane & 15), kh);
        v16bf b1 = load_bfrag_lds(bufB, n0w + 16 + (lane & 15), kh);
        v16bf b2 = load_bfrag_lds(bufB, n0w + 32 + (lane & 15), kh);
        v16bf b3 = load_bfrag_lds(bufB, n0w + 48 + (lane & 15), kh);

        acc0 = __builtin_amdgcn_wmma_f32_16x16x32_bf16(false, af, false, b0, (short)0, acc0, false, false);
        acc1 = __builtin_amdgcn_wmma_f32_16x16x32_bf16(false, af, false, b1, (short)0, acc1, false, false);
        acc2 = __builtin_amdgcn_wmma_f32_16x16x32_bf16(false, af, false, b2, (short)0, acc2, false, false);
        acc3 = __builtin_amdgcn_wmma_f32_16x16x32_bf16(false, af, false, b3, (short)0, acc3, false, false);

        af = afn;
    }

    // epilogue: D layout = lane L holds col (L&15), rows (L>>4)*8 + v
    const int ncol0 = nblk + n0w + (lane & 15);
    const int mrow0 = m0 + (lane >> 4) * 8;
    v8f accs[4] = {acc0, acc1, acc2, acc3};
#pragma unroll
    for (int bi = 0; bi < 4; ++bi) {
        const int ncol = ncol0 + bi * 16;
        const float bv = bias[ncol];
#pragma unroll
        for (int v = 0; v < 8; ++v) {
            float val = accs[bi][v] + bv;
            if (RELU) val = fmaxf(val, 0.f);
            const size_t o = (size_t)(mrow0 + v) * N + ncol;
            if (OUT_BF16) ((__bf16*)Cout)[o] = f2bf(val);
            else          ((float*)Cout)[o]  = val;
        }
    }
#undef STAGE_B
#undef WAIT_ASYNC
}

// ---------------------------------------------------------------------------
// Kernel 7: row softmax over 512. One wave per row; lane holds 16 strided
// elements.
// ---------------------------------------------------------------------------
__global__ void softmax_kernel(const float* __restrict__ H, float* __restrict__ out) {
    const int lane = threadIdx.x & 31;
    const int wave = threadIdx.x >> 5;
    const int b = blockIdx.x * 8 + wave;
    if (b >= BROWS) return;

    const float* r = H + (size_t)b * H3DIM;
    float v[16];
    float mx = -3.4e38f;
#pragma unroll
    for (int i = 0; i < 16; ++i) { v[i] = r[lane + i * 32]; mx = fmaxf(mx, v[i]); }
    for (int m = 16; m > 0; m >>= 1) mx = fmaxf(mx, __shfl_xor(mx, m));
    float s = 0.f;
#pragma unroll
    for (int i = 0; i < 16; ++i) { v[i] = __expf(v[i] - mx); s += v[i]; }
    for (int m = 16; m > 0; m >>= 1) s += __shfl_xor(s, m);
    const float inv = 1.f / s;
    float* o = out + (size_t)b * H3DIM;
#pragma unroll
    for (int i = 0; i < 16; ++i) o[lane + i * 32] = v[i] * inv;
}

// ---------------------------------------------------------------------------
// Launch
// ---------------------------------------------------------------------------
extern "C" void kernel_launch(void* const* d_in, const int* in_sizes, int n_in,
                              void* d_out, int out_size, void* d_ws, size_t ws_size,
                              hipStream_t stream) {
    const float* state   = (const float*)d_in[0];
    const float* now_emb = (const float*)d_in[1];
    const float* neigh   = (const float*)d_in[2];
    // d_in[3] seg_ids: implicit (e = b*DEG + j), unused
    const float* attn_w  = (const float*)d_in[4];
    const float* attn_b  = (const float*)d_in[5];
    const float* aw_w    = (const float*)d_in[6];
    const float* aw_b    = (const float*)d_in[7];
    const float* w1      = (const float*)d_in[8];
    const float* b1      = (const float*)d_in[9];
    const float* w2      = (const float*)d_in[10];
    const float* b2      = (const float*)d_in[11];
    const float* w3      = (const float*)d_in[12];
    const float* b3      = (const float*)d_in[13];

    char* ws = (char*)d_ws;
    size_t off = 0;
    float* vnow = (float*)(ws + off); off += DDIM * sizeof(float);           // 1 KB
    float* vnb  = (float*)(ws + off); off += DDIM * sizeof(float);           // 1 KB
    float* ctot = (float*)(ws + off); off += 256;                            // pad
    __bf16* w1t = (__bf16*)(ws + off); off += (size_t)H1DIM * H0    * 2;
    __bf16* w2t = (__bf16*)(ws + off); off += (size_t)H2DIM * H1DIM * 2;
    __bf16* w3t = (__bf16*)(ws + off); off += (size_t)H3DIM * H2DIM * 2;
    __bf16* X   = (__bf16*)(ws + off); off += (size_t)BROWS * H0    * 2;
    __bf16* H1b = (__bf16*)(ws + off); off += (size_t)BROWS * H1DIM * 2;
    __bf16* H2b = (__bf16*)(ws + off); off += (size_t)BROWS * H2DIM * 2;
    float*  H3f = (float*) (ws + off); off += (size_t)BROWS * H3DIM * 4;
    (void)ws_size; (void)in_sizes; (void)n_in; (void)out_size;

    // 1) fold attention vectors
    prep_attn_kernel<<<1, 256, 0, stream>>>(attn_w, attn_b, aw_w, aw_b, vnow, vnb, ctot);

    // 2) transpose+convert MLP weights to bf16 N x K
    conv_wt_kernel<<<(H0 * H1DIM + 255) / 256, 256, 0, stream>>>(w1, w1t, H0, H1DIM);
    conv_wt_kernel<<<(H1DIM * H2DIM + 255) / 256, 256, 0, stream>>>(w2, w2t, H1DIM, H2DIM);
    conv_wt_kernel<<<(H2DIM * H3DIM + 255) / 256, 256, 0, stream>>>(w3, w3t, H2DIM, H3DIM);

    // 3) attention + aggregation + concat -> X (bf16 8192 x 768)
    attn_kernel<<<BROWS / 8, 256, 0, stream>>>(state, now_emb, neigh, vnow, vnb, ctot, X);

    // 4-6) MLP via bf16 WMMA (async-LDS staged, double buffered)
    gemm_wmma_kernel<true,  true ><<<dim3(BROWS / 32, H1DIM / 256), 256, 0, stream>>>(
        X,   w1t, b1, (void*)H1b, BROWS, H1DIM, H0);
    gemm_wmma_kernel<true,  true ><<<dim3(BROWS / 32, H2DIM / 256), 256, 0, stream>>>(
        H1b, w2t, b2, (void*)H2b, BROWS, H2DIM, H1DIM);
    gemm_wmma_kernel<false, false><<<dim3(BROWS / 32, H3DIM / 256), 256, 0, stream>>>(
        H2b, w3t, b3, (void*)H3f, BROWS, H3DIM, H2DIM);

    // 7) softmax -> d_out (fp32)
    softmax_kernel<<<BROWS / 8, 256, 0, stream>>>(H3f, (float*)d_out);
}
